// LstmRnn_20521353741038
// MI455X (gfx1250) — matrix operand order repaired and checked by reading, not compiled, not run on hardware
//
#include <hip/hip_runtime.h>

typedef __attribute__((ext_vector_type(16))) _Float16 v16h;
typedef __attribute__((ext_vector_type(8)))  _Float16 v8h;
typedef __attribute__((ext_vector_type(8)))  float    v8f;

#define B_        4096
#define T_IN      256
#define F_        32
#define U_        64
#define G_        256   // 4*U
#define OUTSTEPS  24
#define WAVES     4
#define BLOCK     (WAVES * 32)   // 128 threads

__device__ __forceinline__ float rcp_(float x) { return __builtin_amdgcn_rcpf(x); }
__device__ __forceinline__ float sig_(float x) {
    return rcp_(1.0f + __expf(-x));                 // 1 exp + 1 rcp
}
__device__ __forceinline__ float tanh_(float x) {
    float e = __expf(-2.0f * x);                    // 1 exp + 1 rcp
    return (1.0f - e) * rcp_(1.0f + e);
}

// Build a 16x32 f16 A-fragment from an LDS tile (rowlen halfs per row).
// Layout: lanes 0-15 hold M=lane, K = {0..15}; lanes 16-31 hold
// M=lane-16, K = {16..31}.  Both 8-half chunks are contiguous 16B.
__device__ __forceinline__ v16h load_a(const _Float16* buf, int rowlen,
                                       int M, int kOff, int kbA) {
    v8h lo = *(const v8h*)(buf + M * rowlen + kOff + 2 * kbA);
    v8h hi = *(const v8h*)(buf + M * rowlen + kOff + 2 * kbA + 8);
    v16h a;
#pragma unroll
    for (int i = 0; i < 8; ++i) { a[i] = lo[i]; a[8 + i] = hi[i]; }
    return a;
}

#define WMMA(A, Bf, C) \
    __builtin_amdgcn_wmma_f32_16x16x32_f16(false, (A), false, (Bf), (short)0, (C), false, false)

__launch_bounds__(BLOCK)
__global__ void lstm_forecast_kernel(
    const float* __restrict__ x,
    const float* __restrict__ W1, const float* __restrict__ U1, const float* __restrict__ b1,
    const float* __restrict__ W2, const float* __restrict__ U2, const float* __restrict__ b2,
    const float* __restrict__ Wd, const float* __restrict__ bd,
    float* __restrict__ out)
{
    __shared__ __align__(16) _Float16 xbuf[16 * F_];   // x_t tile, f16
    __shared__ __align__(16) _Float16 pbuf[16 * F_];   // pred tile, f16 (decode input)
    __shared__ __align__(16) _Float16 hbuf[16 * U_];   // h state, f16 (A-matrix feed)

    const int tid  = threadIdx.x;
    const int lane = tid & 31;
    const int w    = tid >> 5;                 // wave id 0..3 -> owns units [16w,16w+16) for ALL 4 gates
    const int nCol = lane & 15;
    const int u0   = w * 16;
    const int Mrow = lane & 15;                // A/C row this lane touches
    const int mOff = (lane < 16) ? 0 : 8;      // C/D fragment row offset
    const int kbA  = (lane < 16) ? 0 : 8;      // A-fragment K base
    const int kbB  = (lane < 16) ? 0 : 16;     // B-fragment K base
    const int batch0 = blockIdx.x * 16;

    // ---- phase-switched weight fragments (per wave: 4 gates of one u-slice) ----
    v16h bW[4], bUa[4], bUb[4];
    float bias[4];
    auto load_phase = [&](const float* Wp, const float* Up, const float* bp) {
#pragma unroll
        for (int g = 0; g < 4; ++g) {
            const int col = g * 64 + u0 + nCol;   // gate-g column for this lane
#pragma unroll
            for (int h = 0; h < 16; ++h) {
                const int k = kbB + h;
                bW[g][h]  = (_Float16)Wp[k * G_ + col];
                bUa[g][h] = (_Float16)Up[k * G_ + col];
                bUb[g][h] = (_Float16)Up[(32 + k) * G_ + col];
            }
            bias[g] = bp[col];
        }
    };

    // ---- init state: c in registers (8 rows per lane), h in LDS ----
    float creg[8];
#pragma unroll
    for (int r = 0; r < 8; ++r) creg[r] = 0.f;
    for (int i = tid; i < 16 * U_; i += BLOCK) hbuf[i] = (_Float16)0.f;
    __syncthreads();

    // One LSTM step: gate GEMMs (12 WMMA / wave) + in-register cell update.
    auto step = [&](const _Float16* inbuf) {
        v16h ax  = load_a(inbuf, F_, Mrow, 0, kbA);
        v16h ah0 = load_a(hbuf,  U_, Mrow, 0, kbA);
        v16h ah1 = load_a(hbuf,  U_, Mrow, 32, kbA);
        v8f z[4];
#pragma unroll
        for (int g = 0; g < 4; ++g) {
            v8f a;
#pragma unroll
            for (int r = 0; r < 8; ++r) a[r] = bias[g];   // bias folded into C
            a = WMMA(ax,  bW[g],  a);
            a = WMMA(ah0, bUa[g], a);
            a = WMMA(ah1, bUb[g], a);
            z[g] = a;
        }
        __syncthreads();   // all waves done reading hbuf before overwrite
#pragma unroll
        for (int r = 0; r < 8; ++r) {
            const float iv = sig_(z[0][r]);
            const float fv = sig_(z[1][r]);
            const float gv = tanh_(z[2][r]);
            const float ov = sig_(z[3][r]);
            creg[r] = fv * creg[r] + iv * gv;
            hbuf[(r + mOff) * U_ + u0 + nCol] = (_Float16)(ov * tanh_(creg[r]));
        }
        __syncthreads();   // h visible to all waves
    };

    // ---------------- warmup: 256 steps, x double-buffered through registers ----------------
    load_phase(W1, U1, b1);

    const int xrow = tid >> 5;          // rows 0..3 base; each thread covers 4 rows
    float xreg[4];
#pragma unroll
    for (int q = 0; q < 4; ++q)
        xreg[q] = x[(size_t)(batch0 + xrow + 4 * q) * T_IN * F_ + (lane)];

    for (int t = 0; t < T_IN; ++t) {
        // stage x_t from registers, then immediately issue loads for x_{t+1}
#pragma unroll
        for (int q = 0; q < 4; ++q)
            xbuf[(xrow + 4 * q) * F_ + lane] = (_Float16)xreg[q];
        if (t + 1 < T_IN) {
#pragma unroll
            for (int q = 0; q < 4; ++q) {
                const size_t gix =
                    (size_t)(batch0 + xrow + 4 * q) * T_IN * F_ + (size_t)(t + 1) * F_ + lane;
                xreg[q] = x[gix];
                if (t + 8 < T_IN)
                    __builtin_prefetch(&x[gix + 7 * F_], 0, 3);
            }
        }
        __syncthreads();
        step(xbuf);
    }

    // ---------------- decode: swap weights, 24 autoregressive predictions ----------------
    load_phase(W2, U2, b2);

    v16h bWdA = {}, bWdB = {};
    float biasD = 0.f;
    if (w < 2) {   // waves 0,1 own the two 16-wide tiles of the 32-wide dense head
#pragma unroll
        for (int h = 0; h < 16; ++h) {
            const int k = kbB + h;
            bWdA[h] = (_Float16)Wd[k * F_ + w * 16 + nCol];
            bWdB[h] = (_Float16)Wd[(32 + k) * F_ + w * 16 + nCol];
        }
        biasD = bd[w * 16 + nCol];
    }

    auto do_pred = [&](int s) {
        if (w < 2) {
            v16h ah0 = load_a(hbuf, U_, Mrow, 0, kbA);
            v16h ah1 = load_a(hbuf, U_, Mrow, 32, kbA);
            v8f p;
#pragma unroll
            for (int r = 0; r < 8; ++r) p[r] = biasD;
            p = WMMA(ah0, bWdA, p);
            p = WMMA(ah1, bWdB, p);
#pragma unroll
            for (int r = 0; r < 8; ++r) {
                pbuf[(r + mOff) * F_ + w * 16 + nCol] = (_Float16)p[r];
                out[(size_t)(batch0 + r + mOff) * OUTSTEPS * F_ + s * F_ + w * 16 + nCol] = p[r];
            }
        }
        __syncthreads();
    };

    do_pred(0);
    for (int s = 1; s < OUTSTEPS; ++s) {
        step(pbuf);
        do_pred(s);
    }
}

extern "C" void kernel_launch(void* const* d_in, const int* in_sizes, int n_in,
                              void* d_out, int out_size, void* d_ws, size_t ws_size,
                              hipStream_t stream) {
    (void)in_sizes; (void)n_in; (void)d_ws; (void)ws_size; (void)out_size;
    const float* x  = (const float*)d_in[0];
    const float* W1 = (const float*)d_in[1];
    const float* U1 = (const float*)d_in[2];
    const float* b1 = (const float*)d_in[3];
    const float* W2 = (const float*)d_in[4];
    const float* U2 = (const float*)d_in[5];
    const float* b2 = (const float*)d_in[6];
    const float* Wd = (const float*)d_in[7];
    const float* bd = (const float*)d_in[8];
    float* out = (float*)d_out;

    dim3 grid(B_ / 16);
    dim3 block(BLOCK);
    hipLaunchKernelGGL(lstm_forecast_kernel, grid, block, 0, stream,
                       x, W1, U1, b1, W2, U2, b2, Wd, bd, out);
}
